// Linear_73495480369501
// MI455X (gfx1250) — matrix operand order repaired and checked by reading, not compiled, not run on hardware
//
#include <hip/hip_runtime.h>

// ---------------------------------------------------------------------------
// e3nn Linear (256x0e + 128x1o + 64x2e), N=200000, DIM=960, f32 in/out.
// Strategy: per-irrep GEMM on v_wmma_f32_16x16x32_f16 with split-f16 error
// compensation (hi*hi + lo*hi + hi*lo), f32 accumulate. Weights are
// pre-swizzled (+path-weight folded) into WMMA B-fragment order in d_ws.
// ---------------------------------------------------------------------------

typedef __attribute__((ext_vector_type(16))) _Float16 v16h;
typedef __attribute__((ext_vector_type(8)))  float    v8f;

#define ROWDIM 960
#define TOTW   86016   // 256^2 + 128^2 + 64^2

// ---------------------------------------------------------------------------
// Prep: fold path_weight into W, split into f16 hi/lo, and lay out in
// B-fragment order: [block][wtile][kchunk][lane][slot], where in this code's own
// B(32x16,f16) packing: column n = lane%16, k = 16*(lane/16) + slot.
// ---------------------------------------------------------------------------
__global__ __launch_bounds__(256) void prep_weights(
    const float* __restrict__ w,
    _Float16* __restrict__ whi,
    _Float16* __restrict__ wlo)
{
    const int t = blockIdx.x * blockDim.x + threadIdx.x;
    if (t >= TOTW) return;

    int mul, woff, base; float pw;
    if (t < 65536)      { mul = 256; woff = 0;     base = 0;     pw = 0.0625f;              }
    else if (t < 81920) { mul = 128; woff = 65536; base = 65536; pw = 0.08838834764831845f; }
    else                { mul = 64;  woff = 81920; base = 81920; pw = 0.125f;               }

    const int tp   = t - base;
    const int frag = tp >> 9;        // 512 elements per 32x16 fragment
    const int r    = tp & 511;
    const int lane = r >> 4;
    const int s    = r & 15;
    const int nkc  = mul >> 5;       // K-chunks of 32
    const int wt   = frag / nkc;
    const int kc   = frag - wt * nkc;

    const int k    = kc * 32 + ((lane >> 4) << 4) + s;   // B layout: k = 16*h + slot
    const int wcol = wt * 16 + (lane & 15);              // B layout: n = lane%16

    const float v = pw * w[woff + k * mul + wcol];       // W[u,w], u=k
    const _Float16 hi = (_Float16)v;
    whi[t] = hi;
    wlo[t] = (_Float16)(v - (float)hi);
}

// ---------------------------------------------------------------------------
// Main GEMM per irrep block. Workgroup = 256 threads = 8 waves:
//   wave = rt*4 + wq ; rt in {0,1} row-subtile of 16 atoms, wq in {0..3}
//   wave wq owns w-tiles  wt = wq + 4*j, j < WTPW = MUL/64.
// Rows of the WMMA A-tile are atoms; the irrep component i selects the
// interleaved column (stride D) inside the LDS-staged x rows.
// ---------------------------------------------------------------------------
template<int MUL, int D, int XOFF, int WSOFF, bool BIAS>
__global__ __launch_bounds__(256) void linear_irrep_kernel(
    const float* __restrict__ x,
    const float* __restrict__ bias,
    const _Float16* __restrict__ wh,
    const _Float16* __restrict__ wl,
    float* __restrict__ out)
{
    constexpr int WIDTH = MUL * D;
    constexpr int LDSW  = WIDTH + 4;   // odd*4 bank pattern + float4 alignment
    constexpr int NKC   = MUL / 32;
    constexpr int WTPW  = MUL / 64;

    __shared__ float xs[32 * LDSW];

    const int tid  = threadIdx.x;
    const int wave = tid >> 5;
    const int lane = tid & 31;
    const int rt   = wave >> 2;
    const int wq   = wave & 3;
    const long base_row = (long)blockIdx.x * 32;

    // ---- stage 32 atom-rows of this irrep block into LDS (coalesced) ----
    constexpr int NV4 = 32 * WIDTH / 4;            // divisible by 256 for all blocks
    for (int idx = tid; idx < NV4; idx += 256) {
        const int a = idx / (WIDTH / 4);
        const int c = idx - a * (WIDTH / 4);
        const float4 v = *(const float4*)(x + (base_row + a) * ROWDIM + XOFF + c * 4);
        *(float4*)(&xs[a * LDSW + c * 4]) = v;
    }
    __syncthreads();

    const int mrow = lane & 15;        // A layout: row m = lane%16
    const int half = lane >> 4;
    const float* rowp = &xs[(rt * 16 + mrow) * LDSW];

    v8f C[WTPW][D];
#pragma unroll
    for (int j = 0; j < WTPW; ++j)
#pragma unroll
        for (int i = 0; i < D; ++i) {
            v8f z = {0.f, 0.f, 0.f, 0.f, 0.f, 0.f, 0.f, 0.f};
            C[j][i] = z;
        }

    for (int kc = 0; kc < NKC; ++kc) {
        // ---- build split-f16 A fragments for all D components ----
        v16h Ahi[D], Alo[D];
#pragma unroll
        for (int i = 0; i < D; ++i) {
#pragma unroll
            for (int s = 0; s < 16; ++s) {
                // A layout (matches B packing): k = 16*(lane/16) + s
                const int k = kc * 32 + (half << 4) + s;
                const float v = rowp[k * D + i];
                const _Float16 hi = (_Float16)v;
                Ahi[i][s] = hi;
                Alo[i][s] = (_Float16)(v - (float)hi);
            }
        }
        // ---- B fragments (pre-swizzled, L2-hot), 3 WMMAs per (wt,i) ----
#pragma unroll
        for (int j = 0; j < WTPW; ++j) {
            const int wt  = wq + 4 * j;
            const int off = WSOFF + ((wt * NKC + kc) << 9) + (lane << 4);
            const v16h Bhi = *(const v16h*)(wh + off);
            const v16h Blo = *(const v16h*)(wl + off);
#pragma unroll
            for (int i = 0; i < D; ++i) {
                v8f acc = C[j][i];
                acc = __builtin_amdgcn_wmma_f32_16x16x32_f16(false, Ahi[i], false, Bhi,
                                                             (short)0, acc, false, false);
                acc = __builtin_amdgcn_wmma_f32_16x16x32_f16(false, Alo[i], false, Bhi,
                                                             (short)0, acc, false, false);
                acc = __builtin_amdgcn_wmma_f32_16x16x32_f16(false, Ahi[i], false, Blo,
                                                             (short)0, acc, false, false);
                C[j][i] = acc;
            }
        }
    }

    // ---- write out: D layout col n = lane%16, row m = v + 8*(lane/16) ----
    const int ncol = lane & 15;
#pragma unroll
    for (int j = 0; j < WTPW; ++j) {
        const int wt = wq + 4 * j;
        const float bv = BIAS ? bias[wt * 16 + ncol] : 0.f;
#pragma unroll
        for (int i = 0; i < D; ++i) {
#pragma unroll
            for (int v = 0; v < 8; ++v) {
                const long row = base_row + rt * 16 + v + 8 * half;
                out[row * ROWDIM + XOFF + (long)(wt * 16 + ncol) * D + i] = C[j][i][v] + bv;
            }
        }
    }
}

// ---------------------------------------------------------------------------
extern "C" void kernel_launch(void* const* d_in, const int* in_sizes, int n_in,
                              void* d_out, int out_size, void* d_ws, size_t ws_size,
                              hipStream_t stream)
{
    const float* x = (const float*)d_in[0];
    const float* w = (const float*)d_in[1];
    const float* b = (const float*)d_in[2];
    float* out = (float*)d_out;

    _Float16* wh = (_Float16*)d_ws;          // needs 2*2*86016 = 344 KB of d_ws
    _Float16* wl = wh + TOTW;

    const int n = in_sizes[0] / ROWDIM;      // 200000 atoms
    const int ntiles = n / 32;               // 6250 (exact)

    prep_weights<<<(TOTW + 255) / 256, 256, 0, stream>>>(w, wh, wl);

    // 0e: MUL=256, D=1, xoff=0,   ws base 0,     with bias
    linear_irrep_kernel<256, 1, 0,   0,     true ><<<ntiles, 256, 0, stream>>>(x, b, wh, wl, out);
    // 1o: MUL=128, D=3, xoff=256, ws base 65536
    linear_irrep_kernel<128, 3, 256, 65536, false><<<ntiles, 256, 0, stream>>>(x, b, wh, wl, out);
    // 2e: MUL=64,  D=5, xoff=640, ws base 81920
    linear_irrep_kernel<64,  5, 640, 81920, false><<<ntiles, 256, 0, stream>>>(x, b, wh, wl, out);
}
